// GraphConvolution_27092653703842
// MI455X (gfx1250) — matrix operand order repaired and checked by reading, not compiled, not run on hardware
//
#include <hip/hip_runtime.h>
#include <hip/hip_bf16.h>

typedef __attribute__((ext_vector_type(16))) __bf16 v16bf;
typedef __attribute__((ext_vector_type(8)))  __bf16 v8bf;
typedef __attribute__((ext_vector_type(8)))  float  v8f;
typedef __attribute__((ext_vector_type(4)))  unsigned v4u;

#define GCN_N_NODES 50000
#define GCN_N_EDGES 800000
#define GCN_IN_F    256
#define GCN_HIDDEN  128

// LDS W layout: 128 rows x 256 bf16, padded by 8 bf16 per row to break bank
// conflicts. Row stride = 264 bf16 = 132 dwords = 528 B (16B-aligned, and
// 132 % 64 == 4 -> consecutive rows land 4 banks apart).
#define WROW_DW 132

__device__ __forceinline__ unsigned pack_bf16(float a, float b) {
    __bf16 lo = (__bf16)a, hi = (__bf16)b;
    return (unsigned)__builtin_bit_cast(unsigned short, lo) |
           ((unsigned)__builtin_bit_cast(unsigned short, hi) << 16);
}

// B fragment: lane owns column N (= one W row), 16 consecutive K values
// = 8 LDS dwords starting at `p` (2x ds_load_b128, 16B-aligned).
__device__ __forceinline__ v16bf load_bfrag(const unsigned* p) {
    const v4u q0 = *reinterpret_cast<const v4u*>(p);
    const v4u q1 = *reinterpret_cast<const v4u*>(p + 4);
    const v8bf b0 = __builtin_bit_cast(v8bf, q0);
    const v8bf b1 = __builtin_bit_cast(v8bf, q1);
    return __builtin_shufflevector(b0, b1, 0, 1, 2, 3, 4, 5, 6, 7,
                                   8, 9, 10, 11, 12, 13, 14, 15);
}

// ---------------------------------------------------------------------------
// Kernel 1: support = x @ W^T + b  via v_wmma_f32_16x16x32_bf16.
// Block = 128 threads = 4 waves; block stages W (bf16) in LDS once, then each
// wave computes one 16-node tile x all 128 hidden channels (8 v8f accums).
// K loop: 256/32 = 8 steps, 8 WMMA per step -> 64 WMMA per wave.
// A raw loads are software-pipelined one K-step ahead; B fragments are
// double-buffered so ds_load of ht+1 is in flight during wmma(ht).
// ---------------------------------------------------------------------------
__global__ __launch_bounds__(128) void gcn_gemm_wmma(
    const float* __restrict__ x,     // [N_NODES, IN_F] row-major
    const float* __restrict__ W,     // [HIDDEN, IN_F] row-major
    const float* __restrict__ bias,  // [HIDDEN]
    float* __restrict__ support)     // [N_NODES, HIDDEN]
{
    __shared__ unsigned Wlds[GCN_HIDDEN * WROW_DW];   // 67,584 bytes

    // ---- cooperative W fp32 -> bf16 stage (coalesced float4 reads) ----
#pragma unroll 4
    for (int j = 0; j < 64; ++j) {
        const int q   = threadIdx.x + j * 128;   // quad index
        const int row = q >> 6;                  // 64 quads per row
        const int cq  = q & 63;
        const float4 f = reinterpret_cast<const float4*>(W)[q];
        Wlds[row * WROW_DW + cq * 2 + 0] = pack_bf16(f.x, f.y);
        Wlds[row * WROW_DW + cq * 2 + 1] = pack_bf16(f.z, f.w);
    }
    __syncthreads();

    const int lane = threadIdx.x & 31;
    const int wave = threadIdx.x >> 5;
    const int tile = blockIdx.x * 4 + wave;        // 16-node tile index
    if (tile * 16 >= GCN_N_NODES) return;          // wave-uniform (after barrier)

    const int  m  = lane & 15;                     // row (A) / col (B,C) in tile
    const bool hi = lane >= 16;

    // A fragment source: row m of the node tile; lanes 0-15 own K {0..15},
    // lanes 16-31 own K {16..31} of each 32-wide K step.
    const float4* xp = reinterpret_cast<const float4*>(
        x + (size_t)(tile * 16 + m) * GCN_IN_F + (hi ? 16 : 0));

    // Per-lane base into LDS W: column N=m, K-half picked by `hi`.
    const unsigned* wlane = &Wlds[m * WROW_DW + (hi ? 8 : 0)];

    // Accumulators initialized with bias (bias depends only on column N=m).
    v8f acc[8];
#pragma unroll
    for (int ht = 0; ht < 8; ++ht) {
        const float bv = bias[ht * 16 + m];
#pragma unroll
        for (int r = 0; r < 8; ++r) acc[ht][r] = bv;
    }

    // Prime A pipeline: raw fp32 chunks for kk=0.
    float4 r0 = xp[0], r1 = xp[1], r2 = xp[2], r3 = xp[3];

#pragma unroll
    for (int kk = 0; kk < GCN_IN_F / 32; ++kk) {
        // Convert current A fragment.
        v16bf a;
        a[0]  = (__bf16)r0.x; a[1]  = (__bf16)r0.y;
        a[2]  = (__bf16)r0.z; a[3]  = (__bf16)r0.w;
        a[4]  = (__bf16)r1.x; a[5]  = (__bf16)r1.y;
        a[6]  = (__bf16)r1.z; a[7]  = (__bf16)r1.w;
        a[8]  = (__bf16)r2.x; a[9]  = (__bf16)r2.y;
        a[10] = (__bf16)r2.z; a[11] = (__bf16)r2.w;
        a[12] = (__bf16)r3.x; a[13] = (__bf16)r3.y;
        a[14] = (__bf16)r3.z; a[15] = (__bf16)r3.w;

        // Issue next K-step's global loads so they overlap the WMMA chain.
        if (kk < GCN_IN_F / 32 - 1) {
            const int nb = (kk + 1) * 8;
            r0 = xp[nb + 0]; r1 = xp[nb + 1];
            r2 = xp[nb + 2]; r3 = xp[nb + 3];
        }

        // Double-buffered B fragments.
        v16bf bcur = load_bfrag(wlane + kk * 16);
#pragma unroll
        for (int ht = 0; ht < 8; ++ht) {
            v16bf bnxt;
            if (ht < 7)
                bnxt = load_bfrag(wlane + (ht + 1) * 16 * WROW_DW + kk * 16);
            acc[ht] = __builtin_amdgcn_wmma_f32_16x16x32_bf16(
                /*neg_a=*/false, a, /*neg_b=*/false, bcur,
                /*c_mod=*/(short)0, acc[ht],
                /*reuse_a=*/false, /*reuse_b=*/false);
            bcur = bnxt;
        }
    }

    // Store: VGPR r <-> row (r + (hi?8:0)), column m of each h-tile.
    const int row_base = tile * 16 + (hi ? 8 : 0);
#pragma unroll
    for (int ht = 0; ht < 8; ++ht) {
#pragma unroll
        for (int r = 0; r < 8; ++r) {
            support[(size_t)(row_base + r) * GCN_HIDDEN + ht * 16 + m] = acc[ht][r];
        }
    }
}

// ---------------------------------------------------------------------------
// Kernel 2: zero-init output (d_out is poisoned by the harness)
// ---------------------------------------------------------------------------
__global__ __launch_bounds__(256) void gcn_zero(float4* __restrict__ out, int n4)
{
    int i = blockIdx.x * blockDim.x + threadIdx.x;
    if (i < n4) out[i] = make_float4(0.f, 0.f, 0.f, 0.f);
}

// ---------------------------------------------------------------------------
// Kernel 3: edge scatter — one wave per edge; edge index forced to SGPR so
// adj_row/col/val come in as scalar loads. Lane owns 4 contiguous hidden
// channels: float4 gather + 4x hardware global_atomic_add_f32.
// ---------------------------------------------------------------------------
__global__ __launch_bounds__(256) void gcn_scatter(
    const int* __restrict__ adj_row,
    const int* __restrict__ adj_col,
    const float* __restrict__ adj_val,
    const float* __restrict__ support,
    float* __restrict__ out)
{
    int e = blockIdx.x * 8 + (threadIdx.x >> 5);   // wave-uniform edge index
    e = __builtin_amdgcn_readfirstlane(e);         // force SGPR -> s_load path
    if (e >= GCN_N_EDGES) return;

    const int lane = threadIdx.x & 31;
    const int   r = adj_row[e];
    const int   c = adj_col[e];
    const float v = adj_val[e];

    const float4 g =
        reinterpret_cast<const float4*>(support + (size_t)c * GCN_HIDDEN)[lane];
    float* dst = out + (size_t)r * GCN_HIDDEN + lane * 4;

    unsafeAtomicAdd(dst + 0, g.x * v);   // global_atomic_add_f32 (no CAS loop)
    unsafeAtomicAdd(dst + 1, g.y * v);
    unsafeAtomicAdd(dst + 2, g.z * v);
    unsafeAtomicAdd(dst + 3, g.w * v);
}

// ---------------------------------------------------------------------------
extern "C" void kernel_launch(void* const* d_in, const int* in_sizes, int n_in,
                              void* d_out, int out_size, void* d_ws, size_t ws_size,
                              hipStream_t stream)
{
    const float* x       = (const float*)d_in[0];
    const int*   adj_row = (const int*)  d_in[1];
    const int*   adj_col = (const int*)  d_in[2];
    const float* adj_val = (const float*)d_in[3];
    const float* W       = (const float*)d_in[4];
    const float* bias    = (const float*)d_in[5];
    float*       out     = (float*)d_out;
    float*       support = (float*)d_ws;   // 50000*128*4 = 25.6 MB scratch

    // GEMM: 3125 node tiles, 4 waves (tiles) per 128-thread block.
    const int tiles  = (GCN_N_NODES + 15) / 16;
    const int blocks = (tiles + 3) / 4;
    gcn_gemm_wmma<<<blocks, 128, 0, stream>>>(x, W, bias, support);

    // Zero output (same stream orders it before scatter).
    const int n4 = (GCN_N_NODES * GCN_HIDDEN) / 4;
    gcn_zero<<<(n4 + 255) / 256, 256, 0, stream>>>((float4*)out, n4);

    // Scatter: one wave per edge, 8 edges per 256-thread block.
    const int sblocks = (GCN_N_EDGES + 7) / 8;
    gcn_scatter<<<sblocks, 256, 0, stream>>>(adj_row, adj_col, adj_val, support, out);
}
